// TemporalAttention_55722905699031
// MI455X (gfx1250) — compile-verified, matrix operand order checked by reading, not run on hardware
//
#include <hip/hip_runtime.h>
#include <math.h>

#define DIN 512
#define DT 64
#define DKV 576           // DIN + DT
#define NH 8
#define HD 64
#define BB 32
#define NN 4096
#define NT 32             // n-rows per workgroup tile
#define KSTEPS 18         // 576 / 32 (even -> clean 2x ping-pong)
#define LDS_STRIDE 584    // 576 + 8 halves padding (conflict-free b128 reads)
#define WFRAG_HALF (KSTEPS * 32 * 512)   // halves per swizzled weight matrix = 294912

typedef _Float16 v8h  __attribute__((ext_vector_type(8)));
typedef _Float16 v16h __attribute__((ext_vector_type(16)));
typedef float    v8f  __attribute__((ext_vector_type(8)));

#if defined(__has_builtin) && __has_builtin(__builtin_amdgcn_sched_barrier)
#define SCHED_FENCE() __builtin_amdgcn_sched_barrier(0)
#else
#define SCHED_FENCE() ((void)0)
#endif

// ---------------------------------------------------------------------------
// Convert Wk / Wv (f32 [576,512] row-major) into f16 fragments laid out exactly
// as the WMMA B operand expects: frag = kt*32 + nt; within a frag, lane L's 16
// halves are contiguous. Element (k,n) of B maps to lane = (n%16) with
// K-halves K = ((h<8)?0:16) + 8*(L>>4) + (h&7)  (16-bit operand layout, ISA 7.12.2).
// ---------------------------------------------------------------------------
__global__ void prep_weights_kernel(const float* __restrict__ Wk,
                                    const float* __restrict__ Wv,
                                    _Float16* __restrict__ dst) {
    int tid = blockIdx.x * blockDim.x + threadIdx.x;
    if (tid >= 2 * WFRAG_HALF) return;
    int sel  = tid / WFRAG_HALF;
    int r    = tid % WFRAG_HALF;
    int frag = r >> 9;           // /512 halves per frag
    int idx  = r & 511;
    int lane = idx >> 4;
    int h    = idx & 15;
    int kt   = frag >> 5;
    int nt   = frag & 31;
    int hi   = lane >> 4;
    int koff = ((h < 8) ? 0 : 16) + 8 * hi + (h & 7);
    int k    = kt * 32 + koff;
    int n    = nt * 16 + (lane & 15);
    const float* src = sel ? Wv : Wk;
    dst[tid] = (_Float16)src[(size_t)k * DIN + n];
}

__global__ void zero_kernel(float* __restrict__ p, int n) {
    int t = blockIdx.x * blockDim.x + threadIdx.x;
    if (t < n) p[t] = 0.f;
}

// Q = query @ Wq + bq  (tiny: 32x512x512)
__global__ __launch_bounds__(256)
void qproj_kernel(const float* __restrict__ query, const float* __restrict__ Wq,
                  const float* __restrict__ bq, float* __restrict__ Qws) {
    __shared__ float qrow[DIN];
    int b = blockIdx.x;
    for (int d = threadIdx.x; d < DIN; d += 256) qrow[d] = query[(size_t)b * DIN + d];
    __syncthreads();
    for (int j = threadIdx.x; j < DIN; j += 256) {
        float s = bq[j];
        for (int d = 0; d < DIN; ++d) s = fmaf(qrow[d], Wq[(size_t)d * DIN + j], s);
        Qws[(size_t)b * DIN + j] = s;
    }
}

// ---------------------------------------------------------------------------
// Fused projection GEMM: [NT,576] (key|value ++ time2vec, f16 in LDS)
//                      x [576,512] (swizzled f16 weights) via v_wmma_f32_16x16x32_f16
// Register ping-pong with explicit scheduling fences: prefetch phases (18
// loads for the other buffer set) and WMMA phases are separated by
// sched_barrier(0), so the scheduler cannot rotate loads down to their
// consumers. Each WMMA group waits on loads issued a full phase earlier.
// IS_SCORE: epilogue dots rows of K with Q -> scores/sqrt(HD)
// else:     epilogue weights rows of V by attn -> atomic f32 accumulate
// ---------------------------------------------------------------------------
template <bool IS_SCORE>
__global__ __launch_bounds__(256)
void fused_proj_kernel(const float* __restrict__ kv,     // key or value [B,N,512]
                       const float* __restrict__ td,     // time_delta   [B,N]
                       const float* __restrict__ wt,
                       const float* __restrict__ bt,
                       const _Float16* __restrict__ W16, // swizzled weight
                       const float* __restrict__ bias,   // bk or bv [512]
                       const float* __restrict__ Qws,    // [B,512]   (scores pass)
                       const float* __restrict__ attn,   // [B,H,N]   (out pass)
                       float* __restrict__ out)          // scores [B,H,N] | acc [B,512]
{
    __shared__ __align__(16) _Float16 At[NT * LDS_STRIDE];
    __shared__ float attn_s[NT][NH];

    const int b   = blockIdx.y;
    const int n0  = blockIdx.x * NT;
    const int tid = threadIdx.x;

    // ---- stage: key/value rows (f32 -> f16) into LDS cols [0,512) ----
    for (int i = tid; i < NT * (DIN / 4); i += 256) {
        int row = i / (DIN / 4);
        int c4  = (i % (DIN / 4)) * 4;
        const float4 v = *(const float4*)(kv + ((size_t)b * NN + n0 + row) * DIN + c4);
        _Float16* p = &At[row * LDS_STRIDE + c4];
        p[0] = (_Float16)v.x; p[1] = (_Float16)v.y;
        p[2] = (_Float16)v.z; p[3] = (_Float16)v.w;
    }
    // ---- stage: time2vec into LDS cols [512,576) ----
    for (int i = tid; i < NT * DT; i += 256) {
        int row = i / DT, j = i % DT;
        float f = td[(size_t)b * NN + n0 + row] * wt[j] + bt[j];
        float e = ((j & 1) == 0) ? sinf(f) : cosf(f);
        At[row * LDS_STRIDE + DIN + j] = (_Float16)e;
    }
    if (!IS_SCORE) {
        for (int i = tid; i < NT * NH; i += 256) {
            int row = i / NH, h = i % NH;
            attn_s[row][h] = attn[((size_t)b * NH + h) * NN + n0 + row];
        }
    }
    __syncthreads();

    // ---- GEMM: 8 waves, each owns a 16x128 slab (8 tiles of 16x16) ----
    const int lane = tid & 31;
    const int w    = tid >> 5;
    const int rt   = w & 1;          // row-tile (rows rt*16 .. +15)
    const int ct0  = (w >> 1) * 8;   // first col-tile of 8
    const int m    = lane & 15;
    const int hi   = lane >> 4;

    v8f acc[8] = {};
    const _Float16* arow  = &At[(rt * 16 + m) * LDS_STRIDE] + 8 * hi;
    const _Float16* bbase = W16 + (size_t)lane * 16 + ((size_t)ct0 << 9);

    // prologue: k-step 0 into buffer set A
    v8h  a0A = *(const v8h*)(arow);
    v8h  a1A = *(const v8h*)(arow + 16);
    v16h bfA[8];
#pragma unroll
    for (int i = 0; i < 8; ++i)
        bfA[i] = *(const v16h*)(bbase + ((size_t)i << 9));

#pragma unroll 1
    for (int kt = 0; kt < KSTEPS; kt += 2) {
        // phase 1: prefetch kt+1 into set B (KSTEPS even -> always valid)
        const int k1 = kt + 1;
        const v8h a0B = *(const v8h*)(arow + k1 * 32);
        const v8h a1B = *(const v8h*)(arow + k1 * 32 + 16);
        v16h bfB[8];
#pragma unroll
        for (int i = 0; i < 8; ++i)
            bfB[i] = *(const v16h*)(bbase + (((size_t)k1 * 32 + i) << 9));
        SCHED_FENCE();

        // phase 2: compute kt from set A (operands loaded a full phase ago)
        {
            v16h a;
#pragma unroll
            for (int e = 0; e < 8; ++e) { a[e] = a0A[e]; a[e + 8] = a1A[e]; }
#pragma unroll
            for (int i = 0; i < 8; ++i)
                acc[i] = __builtin_amdgcn_wmma_f32_16x16x32_f16(
                    false, a, false, bfA[i], (short)0, acc[i], false, false);
        }
        SCHED_FENCE();

        // phase 3: prefetch kt+2 into set A (clamped redundant reload at end)
        const int k2 = (kt + 2 < KSTEPS) ? kt + 2 : kt;
        a0A = *(const v8h*)(arow + k2 * 32);
        a1A = *(const v8h*)(arow + k2 * 32 + 16);
#pragma unroll
        for (int i = 0; i < 8; ++i)
            bfA[i] = *(const v16h*)(bbase + (((size_t)k2 * 32 + i) << 9));
        SCHED_FENCE();

        // phase 4: compute kt+1 from set B
        {
            v16h a;
#pragma unroll
            for (int e = 0; e < 8; ++e) { a[e] = a0B[e]; a[e + 8] = a1B[e]; }
#pragma unroll
            for (int i = 0; i < 8; ++i)
                acc[i] = __builtin_amdgcn_wmma_f32_16x16x32_f16(
                    false, a, false, bfB[i], (short)0, acc[i], false, false);
        }
        SCHED_FENCE();
    }

    if (IS_SCORE) {
        // scores[b,h,n] = (K_row . Q_head) / 8 ; K = gemm + bk
#pragma unroll
        for (int g = 0; g < 2; ++g) {
            float ps[8];
#pragma unroll
            for (int v = 0; v < 8; ++v) ps[v] = 0.f;
#pragma unroll
            for (int i = 4 * g; i < 4 * g + 4; ++i) {
                int colg = (ct0 + i) * 16 + m;
                float q  = Qws[(size_t)b * DIN + colg];
                float bc = bias[colg];
#pragma unroll
                for (int v = 0; v < 8; ++v) ps[v] += (acc[i][v] + bc) * q;
            }
#pragma unroll
            for (int off = 8; off >= 1; off >>= 1)
#pragma unroll
                for (int v = 0; v < 8; ++v) ps[v] += __shfl_xor(ps[v], off, 32);
            if (m == 0) {
                int h = (w >> 1) * 2 + g;
#pragma unroll
                for (int v = 0; v < 8; ++v) {
                    int n = n0 + rt * 16 + v + 8 * hi;
                    out[((size_t)b * NH + h) * NN + n] = ps[v] * 0.125f; // 1/sqrt(64)
                }
            }
        }
    } else {
        // acc[b,col] += sum_rows attn[b,h(col),row] * (V + bv)
#pragma unroll
        for (int i = 0; i < 8; ++i) {
            int colg = (ct0 + i) * 16 + m;
            int h    = (ct0 + i) >> 2;
            float bc = bias[colg];
            float s  = 0.f;
#pragma unroll
            for (int v = 0; v < 8; ++v)
                s += (acc[i][v] + bc) * attn_s[rt * 16 + v + 8 * hi][h];
            s += __shfl_xor(s, 16, 32);
            if (hi == 0) atomicAdd(&out[(size_t)b * DIN + colg], s);
        }
    }
}

// masked softmax over n (in place), nan_to_num for all-masked rows
__global__ __launch_bounds__(256)
void softmax_kernel(float* __restrict__ sc, const unsigned char* __restrict__ mask) {
    __shared__ float red[8];
    int bh = blockIdx.x;
    int b  = bh / NH;
    float* row = sc + (size_t)bh * NN;
    const unsigned char* mrow = mask + (size_t)b * NN;
    int tid = threadIdx.x, lane = tid & 31, w = tid >> 5;

    float mx = -INFINITY;
    for (int n = tid; n < NN; n += 256)
        mx = fmaxf(mx, mrow[n] ? row[n] : -INFINITY);
    for (int off = 16; off >= 1; off >>= 1) mx = fmaxf(mx, __shfl_xor(mx, off, 32));
    if (lane == 0) red[w] = mx;
    __syncthreads();
    mx = red[0];
#pragma unroll
    for (int i = 1; i < 8; ++i) mx = fmaxf(mx, red[i]);
    __syncthreads();

    float sum = 0.f;
    bool any = (mx > -INFINITY);
    for (int n = tid; n < NN; n += 256) {
        float e = (any && mrow[n]) ? expf(row[n] - mx) : 0.f;
        row[n] = e;
        sum += e;
    }
    for (int off = 16; off >= 1; off >>= 1) sum += __shfl_xor(sum, off, 32);
    if (lane == 0) red[w] = sum;
    __syncthreads();
    sum = red[0];
#pragma unroll
    for (int i = 1; i < 8; ++i) sum += red[i];
    float inv = (sum > 0.f) ? 1.f / sum : 0.f;
    for (int n = tid; n < NN; n += 256) row[n] *= inv;
}

// out = accOut @ Wo + bo  (tiny)
__global__ __launch_bounds__(256)
void final_proj_kernel(const float* __restrict__ accOut, const float* __restrict__ Wo,
                       const float* __restrict__ bo, float* __restrict__ out) {
    __shared__ float arow[DIN];
    int b = blockIdx.x;
    for (int d = threadIdx.x; d < DIN; d += 256) arow[d] = accOut[(size_t)b * DIN + d];
    __syncthreads();
    for (int j = threadIdx.x; j < DIN; j += 256) {
        float s = bo[j];
        for (int d = 0; d < DIN; ++d) s = fmaf(arow[d], Wo[(size_t)d * DIN + j], s);
        out[(size_t)b * DIN + j] = s;
    }
}

extern "C" void kernel_launch(void* const* d_in, const int* in_sizes, int n_in,
                              void* d_out, int out_size, void* d_ws, size_t ws_size,
                              hipStream_t stream) {
    const float* query = (const float*)d_in[0];
    const float* key   = (const float*)d_in[1];
    const float* value = (const float*)d_in[2];
    const float* tdel  = (const float*)d_in[3];
    const unsigned char* mask = (const unsigned char*)d_in[4]; // jnp bool = 1 byte
    const float* wt = (const float*)d_in[5];
    const float* bt = (const float*)d_in[6];
    const float* Wq = (const float*)d_in[7];
    const float* bq = (const float*)d_in[8];
    const float* Wk = (const float*)d_in[9];
    const float* bk = (const float*)d_in[10];
    const float* Wv = (const float*)d_in[11];
    const float* bv = (const float*)d_in[12];
    const float* Wo = (const float*)d_in[13];
    const float* bo = (const float*)d_in[14];

    char* ws = (char*)d_ws;
    _Float16* W16    = (_Float16*)(ws);                 // Wk16 ++ Wv16, 2*589824 B
    _Float16* Wk16   = W16;
    _Float16* Wv16   = W16 + WFRAG_HALF;
    float*    Qws    = (float*)(ws + 1179648);          // 32*512*4
    float*    scores = (float*)(ws + 1245184);          // 32*8*4096*4
    float*    accOut = (float*)(ws + 5439488);          // 32*512*4

    prep_weights_kernel<<<dim3((2 * WFRAG_HALF + 255) / 256), 256, 0, stream>>>(Wk, Wv, W16);
    zero_kernel<<<dim3((BB * DIN + 255) / 256), 256, 0, stream>>>(accOut, BB * DIN);
    qproj_kernel<<<dim3(BB), 256, 0, stream>>>(query, Wq, bq, Qws);

    fused_proj_kernel<true><<<dim3(NN / NT, BB), 256, 0, stream>>>(
        key, tdel, wt, bt, Wk16, bk, Qws, nullptr, scores);

    softmax_kernel<<<dim3(BB * NH), 256, 0, stream>>>(scores, mask);

    fused_proj_kernel<false><<<dim3(NN / NT, BB), 256, 0, stream>>>(
        value, tdel, wt, bt, Wv16, bv, nullptr, scores, accOut);

    final_proj_kernel<<<dim3(BB), 256, 0, stream>>>(accOut, Wo, bo, (float*)d_out);
}
